// GCN_24747601559656
// MI455X (gfx1250) — matrix operand order repaired and checked by reading, not compiled, not run on hardware
//
#include <hip/hip_runtime.h>
#include <hip/hip_bf16.h>

// ---------------------------------------------------------------------------
// GCN (2x GCNConv + mean-pool + linear) for MI455X / gfx1250.
// GEMMs use v_wmma_f32_16x16x32_bf16 with fragment-major LDS staging so each
// WMMA operand is fetched with wide ds_load_b128 ops. Edge aggregation uses
// f32 global atomics (12.8MB target buffer lives in the 192MB L2).
// ---------------------------------------------------------------------------

#define HID 64

typedef __attribute__((ext_vector_type(16))) __bf16 v16bf;
typedef __attribute__((ext_vector_type(8)))  float  v8f;

__device__ __forceinline__ unsigned short f2bf(float f) {
  // float -> bf16 bits, round-to-nearest-even
  unsigned int u = __float_as_uint(f);
  u += 0x7FFFu + ((u >> 16) & 1u);
  return (unsigned short)(u >> 16);
}

// ----------------------------------------------------------------- utilities
__global__ void zero_kernel(float* __restrict__ p, long long n) {
  long long i = blockIdx.x * (long long)blockDim.x + threadIdx.x;
  if (i < n) p[i] = 0.0f;
}

// deg[dst] += edge_attr  (self-loop "+1" is folded into dis_kernel)
__global__ void deg_kernel(const int* __restrict__ ei, const float* __restrict__ ea,
                           float* __restrict__ deg, int E) {
  int e = blockIdx.x * blockDim.x + threadIdx.x;
  if (e >= E) return;
  atomicAdd(&deg[ei[E + e]], ea[e]);
}

__global__ void dis_kernel(const float* __restrict__ deg, float* __restrict__ dis, int N) {
  int i = blockIdx.x * blockDim.x + threadIdx.x;
  if (i >= N) return;
  float d = deg[i] + 1.0f;  // self-loop weight 1
  dis[i] = (d > 0.0f) ? rsqrtf(fmaxf(d, 1e-12f)) : 0.0f;
}

// ------------------------------------------------------------- WMMA GEMM
// out[N x 64] = A[N x K] * W[K x 64], bf16 inputs, f32 accumulate.
// Block: 256 threads = 8 waves; block tile 128x64; wave tile 16x64.
// LDS holds operands pre-swizzled into WMMA fragment layout so each fragment
// is one contiguous 32B read per lane.
//
// Fragment layouts (ISA 7.12.2, wave32):
//   A (16x32 bf16): lane = 16*half + m ; element i -> k = i + (i&8) + 8*half
//   B (32x16 bf16): lane = 16*half + n ; element i -> k = i + (i&8) + 8*half
//   D (16x16 f32):  vgpr r -> row = r + 8*half ; col = lane&15
template <int K>
__global__ __launch_bounds__(256)
void gemm_bf16_wmma(const float* __restrict__ A, const float* __restrict__ W,
                    float* __restrict__ out, int N) {
  constexpr int NC = K / 32;  // number of K=32 chunks
  __shared__ alignas(32) unsigned short AsF[NC][8][32][16];  // [chunk][wave][lane][elem]
  __shared__ alignas(32) unsigned short BsF[NC][4][32][16];  // [chunk][ctile][lane][elem]

  const int tid  = threadIdx.x;
  const int wave = tid >> 5;
  const int lane = tid & 31;
  const int blockRow = blockIdx.x * 128;

  // ---- stage X tile (128 x K): coalesced float4 loads, 8B LDS stores ----
  constexpr int XQ = 128 * K / 4;
  for (int q = tid; q < XQ; q += 256) {
    int r  = q / (K / 4);
    int k0 = (q % (K / 4)) * 4;
    int row = blockRow + r;
    float4 v = make_float4(0.f, 0.f, 0.f, 0.f);
    if (row < N) v = *(const float4*)(A + (long long)row * K + k0);
    unsigned int lo = (unsigned int)f2bf(v.x) | ((unsigned int)f2bf(v.y) << 16);
    unsigned int hi = (unsigned int)f2bf(v.z) | ((unsigned int)f2bf(v.w) << 16);
    int chunk = k0 >> 5;
    int kin   = k0 & 31;
    int half  = (kin >> 3) & 1;                 // which lane-half holds this k
    int i0    = (kin & 7) + ((kin >> 4) << 3);  // fragment element index
    int ln    = half * 16 + (r & 15);
    *(uint2*)&AsF[chunk][r >> 4][ln][i0] = make_uint2(lo, hi);
  }

  // ---- stage W tile (K x 64): coalesced float4 loads, 4 u16 LDS stores ----
  constexpr int WQ = K * 16;  // quads of 4 columns
  for (int q = tid; q < WQ; q += 256) {
    int k  = q >> 4;
    int c0 = (q & 15) * 4;
    float4 v = *(const float4*)(W + (long long)k * HID + c0);
    int chunk = k >> 5;
    int half  = (k >> 3) & 1;
    int i     = (k & 7) + (((k >> 4) & 1) << 3);
    int t     = c0 >> 4;
    int n0    = half * 16 + (c0 & 15);
    BsF[chunk][t][n0 + 0][i] = f2bf(v.x);
    BsF[chunk][t][n0 + 1][i] = f2bf(v.y);
    BsF[chunk][t][n0 + 2][i] = f2bf(v.z);
    BsF[chunk][t][n0 + 3][i] = f2bf(v.w);
  }
  __syncthreads();

  // ---- compute: each fragment = one 32B contiguous LDS read ----
  v8f acc[4] = {};
#pragma unroll
  for (int chunk = 0; chunk < NC; ++chunk) {
    v16bf afrag = *(const v16bf*)&AsF[chunk][wave][lane][0];
#pragma unroll
    for (int t = 0; t < 4; ++t) {
      v16bf bfrag = *(const v16bf*)&BsF[chunk][t][lane][0];
      acc[t] = __builtin_amdgcn_wmma_f32_16x16x32_bf16(
          false, afrag, false, bfrag, (short)0, acc[t], false, false);
    }
  }

  // ---- store D ----
  const int half = lane >> 4;
  const int mn   = lane & 15;
#pragma unroll
  for (int t = 0; t < 4; ++t) {
#pragma unroll
    for (int r = 0; r < 8; ++r) {
      int row = blockRow + wave * 16 + r + 8 * half;
      if (row < N) out[(long long)row * HID + t * 16 + mn] = acc[t][r];
    }
  }
}

// ------------------------------------------------- edge gather/scatter-add
// 16 threads per edge, 4 features each (float4 gather, 4 f32 atomics).
__global__ void scatter_kernel(const int* __restrict__ ei, const float* __restrict__ ea,
                               const float* __restrict__ dis, const float* __restrict__ hw,
                               float* __restrict__ agg, int E) {
  long long tid = blockIdx.x * (long long)blockDim.x + threadIdx.x;
  int e = (int)(tid >> 4);
  int l = (int)(tid & 15);
  if (e >= E) return;
  int s = ei[e];
  int d = ei[E + e];
  float nrm = dis[s] * ea[e] * dis[d];
  const float4 v = *(const float4*)(hw + (long long)s * HID + l * 4);
  float* o = agg + (long long)d * HID + l * 4;
  atomicAdd(o + 0, nrm * v.x);
  atomicAdd(o + 1, nrm * v.y);
  atomicAdd(o + 2, nrm * v.z);
  atomicAdd(o + 3, nrm * v.w);
}

// agg[i][f] += dis[i]^2 * hw[i][f] + bias[f]; optional ReLU. (self-loop term)
__global__ void finalize_kernel(float* __restrict__ agg, const float* __restrict__ hw,
                                const float* __restrict__ dis, const float* __restrict__ bias,
                                int N, int doRelu) {
  int idx = blockIdx.x * blockDim.x + threadIdx.x;
  if (idx >= N * HID) return;
  int i = idx >> 6, f = idx & 63;
  float self = dis[i] * dis[i];
  float v = agg[idx] + self * hw[idx] + bias[f];
  if (doRelu) v = fmaxf(v, 0.0f);
  agg[idx] = v;
}

// segment-sum over graph ids + node counts
__global__ void pool_kernel(const float* __restrict__ h, const int* __restrict__ batch,
                            float* __restrict__ psum, float* __restrict__ pcnt, int N) {
  int idx = blockIdx.x * blockDim.x + threadIdx.x;
  if (idx >= N * HID) return;
  int i = idx >> 6, f = idx & 63;
  int b = batch[i];
  atomicAdd(&psum[(long long)b * HID + f], h[idx]);
  if (f == 0) atomicAdd(&pcnt[b], 1.0f);
}

// out[g] = (sum_f psum[g][f]*Wlin[f]) / max(cnt,1) + blin
__global__ void out_kernel(const float* __restrict__ psum, const float* __restrict__ pcnt,
                           const float* __restrict__ Wlin, const float* __restrict__ blin,
                           float* __restrict__ out, int G) {
  int g = blockIdx.x * blockDim.x + threadIdx.x;
  if (g >= G) return;
  float s = 0.0f;
#pragma unroll
  for (int f = 0; f < HID; ++f) s += psum[(long long)g * HID + f] * Wlin[f];
  out[g] = s / fmaxf(pcnt[g], 1.0f) + blin[0];
}

// ---------------------------------------------------------------------------
extern "C" void kernel_launch(void* const* d_in, const int* in_sizes, int n_in,
                              void* d_out, int out_size, void* d_ws, size_t ws_size,
                              hipStream_t stream) {
  const float* x     = (const float*)d_in[0];
  const int*   ei    = (const int*)d_in[1];   // [2, E]
  const float* ea    = (const float*)d_in[2]; // [E]
  const int*   batch = (const int*)d_in[3];   // [N]
  const float* W1    = (const float*)d_in[4];
  const float* b1    = (const float*)d_in[5];
  const float* W2    = (const float*)d_in[6];
  const float* b2    = (const float*)d_in[7];
  const float* Wlin  = (const float*)d_in[8];
  const float* blin  = (const float*)d_in[9];
  float* out = (float*)d_out;

  const int IN_DIM = 128;
  const int N = in_sizes[0] / IN_DIM;
  const int E = in_sizes[2];
  const int G = out_size;

  // workspace layout (floats)
  float* deg  = (float*)d_ws;
  float* dis  = deg + N;
  float* bufA = dis + N;                 // h @ W (WMMA output)
  float* bufB = bufA + (size_t)N * HID;  // aggregated messages / h
  float* psum = bufB + (size_t)N * HID;  // [G, HID]
  float* pcnt = psum + (size_t)G * HID;  // [G]

  const int T = 256;
  auto cdiv = [](long long a, long long b) { return (int)((a + b - 1) / b); };

  // degree + normalization
  zero_kernel<<<cdiv(N, T), T, 0, stream>>>(deg, N);
  zero_kernel<<<cdiv((long long)G * HID + G, T), T, 0, stream>>>(psum, (long long)G * HID + G);
  deg_kernel<<<cdiv(E, T), T, 0, stream>>>(ei, ea, deg, E);
  dis_kernel<<<cdiv(N, T), T, 0, stream>>>(deg, dis, N);

  // conv1: hw = x @ W1 ; aggregate ; + self-loop + b1 ; ReLU
  gemm_bf16_wmma<128><<<cdiv(N, 128), T, 0, stream>>>(x, W1, bufA, N);
  zero_kernel<<<cdiv((long long)N * HID, T), T, 0, stream>>>(bufB, (long long)N * HID);
  scatter_kernel<<<cdiv((long long)E * 16, T), T, 0, stream>>>(ei, ea, dis, bufA, bufB, E);
  finalize_kernel<<<cdiv((long long)N * HID, T), T, 0, stream>>>(bufB, bufA, dis, b1, N, 1);

  // conv2: hw = h1 @ W2 ; aggregate ; + self-loop + b2
  gemm_bf16_wmma<64><<<cdiv(N, 128), T, 0, stream>>>(bufB, W2, bufA, N);
  zero_kernel<<<cdiv((long long)N * HID, T), T, 0, stream>>>(bufB, (long long)N * HID);
  scatter_kernel<<<cdiv((long long)E * 16, T), T, 0, stream>>>(ei, ea, dis, bufA, bufB, E);
  finalize_kernel<<<cdiv((long long)N * HID, T), T, 0, stream>>>(bufB, bufA, dis, b2, N, 0);

  // mean pool + linear head
  pool_kernel<<<cdiv((long long)N * HID, T), T, 0, stream>>>(bufB, batch, psum, pcnt, N);
  out_kernel<<<cdiv(G, T), T, 0, stream>>>(psum, pcnt, Wlin, blin, out, G);
}
